// DriftScene_88270167868070
// MI455X (gfx1250) — compile-verified, matrix operand order checked by reading, not run on hardware
//
#include <hip/hip_runtime.h>

typedef __attribute__((ext_vector_type(16))) _Float16 v16h;
typedef __attribute__((ext_vector_type(8)))  _Float16 v8h;
typedef __attribute__((ext_vector_type(8)))  float    v8f;
typedef __attribute__((ext_vector_type(2)))  float    v2f;

#define ROWS   16384      // B*L
#define DM     512
#define FF     2048
#define QKVN   1536
#define NF     4096       // L*CH flattened
#define NB     512        // B
#define DEPTH  4

// ---------------------------------------------------------------------------
// CDNA5 async global->LDS copy (ASYNCcnt-tracked), b128 per lane
// ---------------------------------------------------------------------------
__device__ __forceinline__ void async_ld_b128(unsigned lds_off, const void* gaddr) {
    asm volatile("global_load_async_to_lds_b128 %0, %1, off"
                 :: "v"(lds_off), "v"(gaddr) : "memory");
}
__device__ __forceinline__ void wait_async0() {
    asm volatile("s_wait_asynccnt 0x0" ::: "memory");
}
__device__ __forceinline__ unsigned lds_off_of(const void* p) {
    return (unsigned)(size_t)p;   // AS(3) generic pointer: low 32 bits = LDS offset
}

// ---------------------------------------------------------------------------
// generic f32 -> f16 cast
// ---------------------------------------------------------------------------
__global__ void cast_f32_f16(const float* __restrict__ s, _Float16* __restrict__ d, long n) {
    long i = (long)blockIdx.x * blockDim.x + threadIdx.x;
    long stride = (long)gridDim.x * blockDim.x;
    for (; i < n; i += stride) d[i] = (_Float16)s[i];
}

// ---------------------------------------------------------------------------
// WMMA GEMM: C[M,N] = A[M,K] * W[N,K]^T + bias[N]   (A,W f16; acc f32)
// Block tile 128x128, 8 waves (2x4), wave tile 64x32 (4x2 wmma tiles).
// Double-buffered LDS filled by async global->LDS; one barrier per k-step.
// Requires M%128==0, N%128==0, K%32==0 (true for all shapes used here).
// ---------------------------------------------------------------------------
enum { OF32 = 1, OF16 = 2, RELUF = 4, RESF = 8 };

template<int F>
__global__ __launch_bounds__(256) void gemm_f16_wmma(
    const _Float16* __restrict__ A, const _Float16* __restrict__ W,
    const float* __restrict__ bias,
    float* __restrict__ Cf, _Float16* __restrict__ Ch,
    const float* __restrict__ resid,
    int M, int N, int K)
{
    __shared__ _Float16 As[2][128][40];
    __shared__ _Float16 Bs[2][128][40];

    const int tid  = threadIdx.x;
    const int lane = tid & 31;
    const int wave = tid >> 5;
    const int wm   = (wave & 1) * 64;   // wave M offset in block
    const int wn   = (wave >> 1) * 32;  // wave N offset in block

    // cooperative tile fill: 2 threads per row, 16 halves (32B = 2 x b128) each
    const int lr = tid >> 1;
    const int lc = (tid & 1) << 4;
    const _Float16* Ag = A + (size_t)(blockIdx.x * 128 + lr) * K + lc;
    const _Float16* Wg = W + (size_t)(blockIdx.y * 128 + lr) * K + lc;
    const unsigned sA0 = lds_off_of(&As[0][lr][lc]);
    const unsigned sA1 = lds_off_of(&As[1][lr][lc]);
    const unsigned sB0 = lds_off_of(&Bs[0][lr][lc]);
    const unsigned sB1 = lds_off_of(&Bs[1][lr][lc]);

    // fragment addressing (ISA 7.12.2, 16-bit A 16x32 layout)
    const int fr = lane & 15;
    const int kb = (lane >> 4) << 3;    // 0 or 8

    v8f acc[4][2];
#pragma unroll
    for (int i = 0; i < 4; i++)
#pragma unroll
        for (int j = 0; j < 2; j++)
#pragma unroll
            for (int e = 0; e < 8; e++) acc[i][j][e] = 0.f;

    auto issue = [&](int k0, int p) {
        const unsigned a = p ? sA1 : sA0;
        const unsigned b = p ? sB1 : sB0;
        async_ld_b128(a,      Ag + k0);
        async_ld_b128(a + 16, Ag + k0 + 8);
        async_ld_b128(b,      Wg + k0);
        async_ld_b128(b + 16, Wg + k0 + 8);
    };

    const int nk = K >> 5;
    issue(0, 0);
    wait_async0();
    __syncthreads();

    for (int kk = 0; kk < nk; kk++) {
        const int p = kk & 1;
        if (kk + 1 < nk) issue((kk + 1) << 5, p ^ 1);

        v16h af[4], bf[2];
#pragma unroll
        for (int i = 0; i < 4; i++) {
            const v8h lo = *(const v8h*)&As[p][wm + i * 16 + fr][kb];
            const v8h hi = *(const v8h*)&As[p][wm + i * 16 + fr][kb + 16];
#pragma unroll
            for (int e = 0; e < 8; e++) { af[i][e] = lo[e]; af[i][e + 8] = hi[e]; }
        }
#pragma unroll
        for (int j = 0; j < 2; j++) {
            const v8h lo = *(const v8h*)&Bs[p][wn + j * 16 + fr][kb];
            const v8h hi = *(const v8h*)&Bs[p][wn + j * 16 + fr][kb + 16];
#pragma unroll
            for (int e = 0; e < 8; e++) { bf[j][e] = lo[e]; bf[j][e + 8] = hi[e]; }
        }
#pragma unroll
        for (int i = 0; i < 4; i++)
#pragma unroll
            for (int j = 0; j < 2; j++)
                acc[i][j] = __builtin_amdgcn_wmma_f32_16x16x32_f16(
                    false, af[i], false, bf[j], (short)0, acc[i][j], false, false);

        if (kk + 1 < nk) {
            wait_async0();      // next tile landed in LDS (in-order per wave)
            __syncthreads();    // ...for all waves; and all reads of next buf done
        }
    }

    // epilogue: C/D layout: VGPR r -> M = r + (lane>>4)*8 ; N = lane&15
    const int r0   = blockIdx.x * 128 + wm;
    const int c0   = blockIdx.y * 128 + wn;
    const int mofs = (lane >> 4) * 8;
#pragma unroll
    for (int i = 0; i < 4; i++)
#pragma unroll
        for (int j = 0; j < 2; j++) {
            const int col = c0 + j * 16 + fr;
            const float bb = bias[col];
#pragma unroll
            for (int r = 0; r < 8; r++) {
                const int row = r0 + i * 16 + mofs + r;
                float v = acc[i][j][r] + bb;
                if (F & RELUF) v = fmaxf(v, 0.f);
                if (F & RESF)  v += resid[(size_t)row * N + col];
                if (F & OF32)  Cf[(size_t)row * N + col] = v;
                if (F & OF16)  Ch[(size_t)row * N + col] = (_Float16)v;
            }
        }
}

// ---------------------------------------------------------------------------
// LayerNorm over 512 cols, one wave32 per row, 8 rows per block; f16 out
// ---------------------------------------------------------------------------
__global__ __launch_bounds__(256) void layernorm_f16(
    const float* __restrict__ x, const float* __restrict__ g,
    const float* __restrict__ b, _Float16* __restrict__ out)
{
    const int wave = threadIdx.x >> 5, lane = threadIdx.x & 31;
    const int row  = blockIdx.x * 8 + wave;
    const float* xr = x + (size_t)row * DM;
    float vals[16];
    float s = 0.f;
#pragma unroll
    for (int i = 0; i < 16; i++) { vals[i] = xr[lane + i * 32]; s += vals[i]; }
#pragma unroll
    for (int off = 16; off > 0; off >>= 1) s += __shfl_xor(s, off, 32);
    const float mean = s * (1.f / 512.f);
    float v2 = 0.f;
#pragma unroll
    for (int i = 0; i < 16; i++) { float d = vals[i] - mean; v2 += d * d; }
#pragma unroll
    for (int off = 16; off > 0; off >>= 1) v2 += __shfl_xor(v2, off, 32);
    const float inv = rsqrtf(v2 * (1.f / 512.f) + 1e-5f);
    _Float16* orow = out + (size_t)row * DM;
#pragma unroll
    for (int i = 0; i < 16; i++) {
        const int c = lane + i * 32;
        orow[c] = (_Float16)((vals[i] - mean) * inv * g[c] + b[c]);
    }
}

// ---------------------------------------------------------------------------
// Attention: one wave per (b,h); L=32 (lane == query row), DH=64
// ---------------------------------------------------------------------------
__global__ __launch_bounds__(32) void attention_k(
    const _Float16* __restrict__ qkv, _Float16* __restrict__ out)
{
    __shared__ _Float16 ks[32][64];
    __shared__ _Float16 vs[32][64];
    const int lane = threadIdx.x;
    const int b = blockIdx.x >> 3, h = blockIdx.x & 7;
    const _Float16* base = qkv + (size_t)b * 32 * QKVN + h * 64;
#pragma unroll
    for (int d = 0; d < 64; d += 8) {
        *(v8h*)&ks[lane][d] = *(const v8h*)(base + (size_t)lane * QKVN + 512 + d);
        *(v8h*)&vs[lane][d] = *(const v8h*)(base + (size_t)lane * QKVN + 1024 + d);
    }
    float q[64];
#pragma unroll
    for (int d = 0; d < 64; d++) q[d] = (float)base[(size_t)lane * QKVN + d];
    __syncthreads();

    float sc[32];
    float mx = -1e30f;
#pragma unroll
    for (int j = 0; j < 32; j++) {
        float s = 0.f;
#pragma unroll
        for (int d = 0; d < 64; d++) s += q[d] * (float)ks[j][d];
        s *= 0.125f;
        sc[j] = s;
        mx = fmaxf(mx, s);
    }
    float den = 0.f;
#pragma unroll
    for (int j = 0; j < 32; j++) { sc[j] = __expf(sc[j] - mx); den += sc[j]; }
    const float rden = 1.f / den;

    float o[64];
#pragma unroll
    for (int d = 0; d < 64; d++) o[d] = 0.f;
#pragma unroll
    for (int j = 0; j < 32; j++) {
        const float p = sc[j];
#pragma unroll
        for (int d = 0; d < 64; d++) o[d] += p * (float)vs[j][d];
    }
    _Float16* orow = out + (size_t)(b * 32 + lane) * DM + h * 64;
#pragma unroll
    for (int d = 0; d < 64; d++) orow[d] = (_Float16)(o[d] * rden);
}

// ---------------------------------------------------------------------------
// row squared-norms
// ---------------------------------------------------------------------------
__global__ __launch_bounds__(256) void row_norm2_k(
    const float* __restrict__ x, float* __restrict__ out, int ncols)
{
    __shared__ float red[256];
    const int row = blockIdx.x;
    float s = 0.f;
    for (int c = threadIdx.x; c < ncols; c += 256) {
        float v = x[(size_t)row * ncols + c];
        s += v * v;
    }
    red[threadIdx.x] = s; __syncthreads();
    for (int off = 128; off > 0; off >>= 1) {
        if (threadIdx.x < off) red[threadIdx.x] += red[threadIdx.x + off];
        __syncthreads();
    }
    if (threadIdx.x == 0) out[row] = red[0];
}

// ---------------------------------------------------------------------------
// build A (512 x 1024): A[i,j<512] = -dist(xf_i,p_j); A[i,512+j] = -dist(xf_i,xf_j)-1e6*(i==j)
// ---------------------------------------------------------------------------
#if __has_builtin(__builtin_amdgcn_wmma_f32_16x16x4_f32)
#define HAVE_WMMA_F32 1
// f32 WMMA version: one wave per 16x16 tile, K-step 4 (A 16x4 = v2f per lane)
__global__ __launch_bounds__(32) void build_A_wmma(
    const float* __restrict__ xf, const float* __restrict__ p,
    const float* __restrict__ xn, const float* __restrict__ pn,
    float* __restrict__ Aout)
{
    const int lane = threadIdx.x;
    const int ti = blockIdx.x, tj = blockIdx.y;
    const int jbase = tj * 16;
    const bool neg = jbase >= 512;
    const float* Y = neg ? xf : p;
    const int jb = neg ? jbase - 512 : jbase;
    const int m  = lane & 15;
    const int kl = (lane >> 4) << 1;   // 0 or 2  (32-bit A 16x4 layout)
    const float* Ar = xf + (size_t)(ti * 16 + m) * NF + kl;
    const float* Br = Y  + (size_t)(jb + m) * NF + kl;

    v8f acc;
#pragma unroll
    for (int e = 0; e < 8; e++) acc[e] = 0.f;
    for (int k = 0; k < NF; k += 4) {
        const v2f a = *(const v2f*)(Ar + k);
        const v2f b = *(const v2f*)(Br + k);
        acc = __builtin_amdgcn_wmma_f32_16x16x4_f32(
            false, a, false, b, (short)0, acc, false, false);
    }
    const int mo = (lane >> 4) * 8;
#pragma unroll
    for (int r = 0; r < 8; r++) {
        const int i  = ti * 16 + mo + r;
        const int jj = jb + (lane & 15);
        const int j  = jbase + (lane & 15);
        const float yn = neg ? xn[jj] : pn[jj];
        const float d2 = xn[i] + yn - 2.f * acc[r];
        float a = -sqrtf(fmaxf(d2, 0.f));
        if (neg && i == jj) a -= 1e6f;
        Aout[(size_t)i * 1024 + j] = a;
    }
}
#else
#define HAVE_WMMA_F32 0
#endif

// scalar fallback (also kept for reference)
__global__ __launch_bounds__(256) void build_A_k(
    const float* __restrict__ xf, const float* __restrict__ p,
    const float* __restrict__ xn, const float* __restrict__ pn,
    float* __restrict__ Aout)
{
    __shared__ float xs[16][68];
    __shared__ float ys[16][68];
    const int ti = blockIdx.x, tj = blockIdx.y;
    const int tid = threadIdx.x;
    const int jbase = tj * 16;
    const bool neg = jbase >= 512;
    const float* Y = neg ? xf : p;
    const int jb = neg ? jbase - 512 : jbase;
    const int lr = tid >> 4;
    const int lc4 = (tid & 15) * 4;
    const int ii = tid >> 4, jj = tid & 15;
    float acc = 0.f;
    for (int k = 0; k < NF; k += 64) {
        __syncthreads();
        *(float4*)&xs[lr][lc4] = *(const float4*)(xf + (size_t)(ti * 16 + lr) * NF + k + lc4);
        *(float4*)&ys[lr][lc4] = *(const float4*)(Y  + (size_t)(jb + lr) * NF + k + lc4);
        __syncthreads();
#pragma unroll
        for (int kk = 0; kk < 64; kk++) acc += xs[ii][kk] * ys[jj][kk];
    }
    const int i = ti * 16 + ii;
    const int j = jbase + jj;
    const float yn = neg ? xn[jb + jj] : pn[jb + jj];
    const float d2 = xn[i] + yn - 2.f * acc;
    float a = -sqrtf(fmaxf(d2, 0.f));
    if (neg && i == jb + jj) a -= 1e6f;
    Aout[(size_t)i * 1024 + j] = a;
}

// ---------------------------------------------------------------------------
// column stats of A; per-row softmax combine + W weights; V-GEMM + loss
// ---------------------------------------------------------------------------
__global__ __launch_bounds__(256) void col_stats_k(
    const float* __restrict__ A, float* __restrict__ colmax, float* __restrict__ colsum)
{
    const int j = blockIdx.x * 256 + threadIdx.x;
    float m = -1e30f;
    for (int i = 0; i < 512; i++) m = fmaxf(m, A[(size_t)i * 1024 + j]);
    float s = 0.f;
    for (int i = 0; i < 512; i++) s += __expf(A[(size_t)i * 1024 + j] - m);
    colmax[j] = m; colsum[j] = s;
}

__global__ __launch_bounds__(256) void row_transform_k(
    float* __restrict__ A, const float* __restrict__ colmax, const float* __restrict__ colsum)
{
    __shared__ float red[256];
    const int row = blockIdx.x, tid = threadIdx.x;
    float* Ar = A + (size_t)row * 1024;
    float a[4];
#pragma unroll
    for (int q = 0; q < 4; q++) a[q] = Ar[tid + q * 256];

    float m = fmaxf(fmaxf(a[0], a[1]), fmaxf(a[2], a[3]));
    red[tid] = m; __syncthreads();
    for (int off = 128; off > 0; off >>= 1) {
        if (tid < off) red[tid] = fmaxf(red[tid], red[tid + off]);
        __syncthreads();
    }
    const float rowmax = red[0]; __syncthreads();

    float e[4]; float s = 0.f;
#pragma unroll
    for (int q = 0; q < 4; q++) { e[q] = __expf(a[q] - rowmax); s += e[q]; }
    red[tid] = s; __syncthreads();
    for (int off = 128; off > 0; off >>= 1) {
        if (tid < off) red[tid] += red[tid + off];
        __syncthreads();
    }
    const float rowsum = red[0]; __syncthreads();

    float asoft[4]; float sp = 0.f, sn = 0.f;
#pragma unroll
    for (int q = 0; q < 4; q++) {
        const int j = tid + q * 256;
        const float acol = __expf(a[q] - colmax[j]) / colsum[j];
        const float v = sqrtf((e[q] / rowsum) * acol);
        asoft[q] = v;
        if (j < 512) sp += v; else sn += v;
    }
    red[tid] = sp; __syncthreads();
    for (int off = 128; off > 0; off >>= 1) {
        if (tid < off) red[tid] += red[tid + off];
        __syncthreads();
    }
    sp = red[0]; __syncthreads();
    red[tid] = sn; __syncthreads();
    for (int off = 128; off > 0; off >>= 1) {
        if (tid < off) red[tid] += red[tid + off];
        __syncthreads();
    }
    sn = red[0]; __syncthreads();

    const float wpos = sn, wneg = sn * sp;
#pragma unroll
    for (int q = 0; q < 4; q++) {
        const int j = tid + q * 256;
        Ar[j] = asoft[q] * (j < 512 ? wpos : wneg);
    }
}

__global__ __launch_bounds__(256) void v_loss_k(
    const float* __restrict__ Wmat, const float* __restrict__ p,
    const float* __restrict__ xf, float* __restrict__ partials)
{
    __shared__ float Wrow[1024];
    __shared__ float red[256];
    const int row = blockIdx.x >> 4;
    const int c = (blockIdx.x & 15) * 256 + threadIdx.x;
    for (int q = threadIdx.x; q < 1024; q += 256) Wrow[q] = Wmat[(size_t)row * 1024 + q];
    __syncthreads();
    float acc = 0.f;
    for (int j = 0; j < 512; j++) acc += Wrow[j] * p[(size_t)j * NF + c];
    for (int j = 0; j < 512; j++) acc -= Wrow[512 + j] * xf[(size_t)j * NF + c];
    red[threadIdx.x] = acc * acc; __syncthreads();
    for (int off = 128; off > 0; off >>= 1) {
        if (threadIdx.x < off) red[threadIdx.x] += red[threadIdx.x + off];
        __syncthreads();
    }
    if (threadIdx.x == 0) partials[blockIdx.x] = red[0];
}

__global__ __launch_bounds__(256) void final_reduce_k(
    const float* __restrict__ partials, float* __restrict__ out)
{
    __shared__ float red[256];
    float s = 0.f;
    for (int i = threadIdx.x; i < 8192; i += 256) s += partials[i];
    red[threadIdx.x] = s; __syncthreads();
    for (int off = 128; off > 0; off >>= 1) {
        if (threadIdx.x < off) red[threadIdx.x] += red[threadIdx.x + off];
        __syncthreads();
    }
    if (threadIdx.x == 0) out[0] = red[0] * (1.f / (512.f * 4096.f));
}

// ---------------------------------------------------------------------------
// host side
// ---------------------------------------------------------------------------
static inline void launch_cast(const float* s, _Float16* d, long n, hipStream_t st) {
    long blocks = (n + 255) / 256;
    if (blocks > 16384) blocks = 16384;
    cast_f32_f16<<<(unsigned)blocks, 256, 0, st>>>(s, d, n);
}

extern "C" void kernel_launch(void* const* d_in, const int* in_sizes, int n_in,
                              void* d_out, int out_size, void* d_ws, size_t ws_size,
                              hipStream_t stream)
{
    const float* sample_p = (const float*)d_in[0];
    const float* eps      = (const float*)d_in[1];
    const float* in_w     = (const float*)d_in[2];
    const float* in_b     = (const float*)d_in[3];
    const float* Wqkv     = (const float*)d_in[4];
    const float* bqkv     = (const float*)d_in[5];
    const float* Wo       = (const float*)d_in[6];
    const float* bo       = (const float*)d_in[7];
    const float* ln1_g    = (const float*)d_in[8];
    const float* ln1_b    = (const float*)d_in[9];
    const float* W1       = (const float*)d_in[10];
    const float* b1       = (const float*)d_in[11];
    const float* W2       = (const float*)d_in[12];
    const float* b2       = (const float*)d_in[13];
    const float* ln2_g    = (const float*)d_in[14];
    const float* ln2_b    = (const float*)d_in[15];
    const float* out_w    = (const float*)d_in[16];
    const float* out_b    = (const float*)d_in[17];

    char* wsp = (char*)d_ws;
    auto carve = [&](size_t bytes) -> void* {
        void* r = (void*)wsp;
        wsp += (bytes + 255) & ~(size_t)255;
        return r;
    };
    float*     x      = (float*)    carve((size_t)ROWS * DM * 4);
    float*     xf     = (float*)    carve((size_t)ROWS * 128 * 4);   // (512,4096)
    _Float16*  h      = (_Float16*) carve((size_t)ROWS * DM * 2);
    _Float16*  big    = (_Float16*) carve((size_t)ROWS * FF * 2);    // qkv / ffn-mid
    _Float16*  eps16  = (_Float16*) carve((size_t)ROWS * 128 * 2);
    _Float16*  inw16  = (_Float16*) carve((size_t)DM * 128 * 2);
    _Float16*  wqkv16 = (_Float16*) carve((size_t)DEPTH * QKVN * DM * 2);
    _Float16*  wo16   = (_Float16*) carve((size_t)DEPTH * DM * DM * 2);
    _Float16*  w116   = (_Float16*) carve((size_t)DEPTH * FF * DM * 2);
    _Float16*  w216   = (_Float16*) carve((size_t)DEPTH * DM * FF * 2);
    _Float16*  outw16 = (_Float16*) carve((size_t)128 * DM * 2);
    float*     Amat   = (float*)    carve((size_t)512 * 1024 * 4);
    float*     xn     = (float*)    carve(512 * 4);
    float*     pn     = (float*)    carve(512 * 4);
    float*     colmax = (float*)    carve(1024 * 4);
    float*     colsum = (float*)    carve(1024 * 4);
    float*     parts  = (float*)    carve(8192 * 4);

    launch_cast(eps,   eps16,  (long)ROWS * 128,        stream);
    launch_cast(in_w,  inw16,  (long)DM * 128,          stream);
    launch_cast(Wqkv,  wqkv16, (long)DEPTH * QKVN * DM, stream);
    launch_cast(Wo,    wo16,   (long)DEPTH * DM * DM,   stream);
    launch_cast(W1,    w116,   (long)DEPTH * FF * DM,   stream);
    launch_cast(W2,    w216,   (long)DEPTH * DM * FF,   stream);
    launch_cast(out_w, outw16, (long)128 * DM,          stream);

    gemm_f16_wmma<OF32><<<dim3(ROWS / 128, DM / 128), 256, 0, stream>>>(
        eps16, inw16, in_b, x, nullptr, nullptr, ROWS, DM, 128);

    for (int i = 0; i < DEPTH; i++) {
        layernorm_f16<<<ROWS / 8, 256, 0, stream>>>(x, ln1_g + i * DM, ln1_b + i * DM, h);
        gemm_f16_wmma<OF16><<<dim3(ROWS / 128, QKVN / 128), 256, 0, stream>>>(
            h, wqkv16 + (size_t)i * QKVN * DM, bqkv + (size_t)i * QKVN,
            nullptr, big, nullptr, ROWS, QKVN, DM);
        attention_k<<<NB * 8, 32, 0, stream>>>(big, h);
        gemm_f16_wmma<OF32 | RESF><<<dim3(ROWS / 128, DM / 128), 256, 0, stream>>>(
            h, wo16 + (size_t)i * DM * DM, bo + (size_t)i * DM,
            x, nullptr, x, ROWS, DM, DM);
        layernorm_f16<<<ROWS / 8, 256, 0, stream>>>(x, ln2_g + i * DM, ln2_b + i * DM, h);
        gemm_f16_wmma<OF16 | RELUF><<<dim3(ROWS / 128, FF / 128), 256, 0, stream>>>(
            h, w116 + (size_t)i * FF * DM, b1 + (size_t)i * FF,
            nullptr, big, nullptr, ROWS, FF, DM);
        gemm_f16_wmma<OF32 | RESF><<<dim3(ROWS / 128, DM / 128), 256, 0, stream>>>(
            big, w216 + (size_t)i * DM * FF, b2 + (size_t)i * DM,
            x, nullptr, x, ROWS, DM, FF);
    }

    launch_cast(x, h, (long)ROWS * DM, stream);
    gemm_f16_wmma<OF32><<<dim3(ROWS / 128, 1), 256, 0, stream>>>(
        h, outw16, out_b, xf, nullptr, nullptr, ROWS, 128, DM);

    row_norm2_k<<<512, 256, 0, stream>>>(xf, xn, NF);
    row_norm2_k<<<512, 256, 0, stream>>>(sample_p, pn, NF);
#if HAVE_WMMA_F32
    build_A_wmma<<<dim3(32, 64), 32, 0, stream>>>(xf, sample_p, xn, pn, Amat);
#else
    build_A_k<<<dim3(32, 64), 256, 0, stream>>>(xf, sample_p, xn, pn, Amat);
#endif
    col_stats_k<<<4, 256, 0, stream>>>(Amat, colmax, colsum);
    row_transform_k<<<512, 256, 0, stream>>>(Amat, colmax, colsum);
    v_loss_k<<<512 * 16, 256, 0, stream>>>(Amat, sample_p, xf, parts);
    final_reduce_k<<<1, 256, 0, stream>>>(parts, (float*)d_out);
}
